// JKNet_27419071218301
// MI455X (gfx1250) — matrix operand order repaired and checked by reading, not compiled, not run on hardware
//
#include <hip/hip_runtime.h>
#include <hip/hip_bf16.h>
#include <math.h>

// ---------------------------------------------------------------------------
// JKNet (6-layer GCN + jumping-knowledge concat) for MI455X / gfx1250.
//  - GEMMs: v_wmma_f32_16x16x32_bf16, N-tile-blocked (one wave -> 16 x 64 strip)
//  - Aggregation: guaranteed-native global_atomic_add_f32 (inline asm, DEV scope)
//  - Whole working set (~90 MB) is resident in the 192 MB global L2.
// ---------------------------------------------------------------------------

typedef __bf16 bf16_t;
typedef __attribute__((ext_vector_type(8)))  bf16_t v8bf;
typedef __attribute__((ext_vector_type(16))) bf16_t v16bf;
typedef __attribute__((ext_vector_type(8)))  float  v8f;

#define NN 50000
#define EE 800000
#define FF 128
#define HH 64
#define LL 6
#define CC 40
#define CP 48          // padded output classes (3 tiles of 16)
#define JK (LL*HH)     // 384

// Native CDNA5 f32 atomic add, device scope (cross-WGP reduction into L2).
__device__ __forceinline__ void gatomic_add_f32(float* p, float v) {
  asm volatile("global_atomic_add_f32 %0, %1, off scope:SCOPE_DEV"
               : : "v"(p), "v"(v) : "memory");
}

// ----------------------------- small utility kernels -----------------------

__global__ void fill_zero_f32(float* p, int n) {
  int i = blockIdx.x * blockDim.x + threadIdx.x;
  if (i < n) p[i] = 0.0f;
}

__global__ void cvt_f32_bf16(const float* __restrict__ src, bf16_t* __restrict__ dst, int n) {
  int i = blockIdx.x * blockDim.x + threadIdx.x;
  if (i < n) dst[i] = (bf16_t)src[i];
}

// Transpose-convert weight: src is [K x Nsrc] row-major f32; dst is [Npad x K]
// row-major bf16 (rows >= Nsrc zero-padded).
__global__ void cvt_wT_bf16(const float* __restrict__ src, bf16_t* __restrict__ dst,
                            int K, int Nsrc, int Npad) {
  int idx = blockIdx.x * blockDim.x + threadIdx.x;
  if (idx >= K * Npad) return;
  int n = idx / K, k = idx - n * K;
  dst[idx] = (n < Nsrc) ? (bf16_t)src[(size_t)k * Nsrc + n] : (bf16_t)0.0f;
}

// ----------------------------- graph normalization -------------------------

__global__ void deg_accum(const int* __restrict__ col, const float* __restrict__ ew,
                          float* __restrict__ deg) {
  int e = blockIdx.x * blockDim.x + threadIdx.x;
  if (e >= EE) return;
  gatomic_add_f32(&deg[col[e]], ew[e]);
}

__global__ void dinv_finalize(float* __restrict__ d) {
  int i = blockIdx.x * blockDim.x + threadIdx.x;
  if (i >= NN) return;
  d[i] = rsqrtf(d[i] + 1.0f);          // +1 = self-loop weight; deg >= 1 always
}

__global__ void norm_compute(const int* __restrict__ row, const int* __restrict__ col,
                             const float* __restrict__ ew, const float* __restrict__ dinv,
                             float* __restrict__ norm) {
  int e = blockIdx.x * blockDim.x + threadIdx.x;
  if (e >= EE) return;
  norm[e] = dinv[row[e]] * ew[e] * dinv[col[e]];
}

// ----------------------------- WMMA GEMM -----------------------------------
// One wave computes a 16 x (NT*16) output strip: the A fragment is loaded once
// per K-step and reused for NT WMMAs (NT accumulators = NT*8 VGPRs).
// Bt is the transposed bf16 weight, stored [Ncols x K] row-major.
// If agg != nullptr, the epilogue also writes agg = D * dinv[row]^2 (self-loop
// initialization of the GCN aggregation, fused to save a full pass over h).
template<int NT>
__global__ void gemm_bf16_wmma(const bf16_t* __restrict__ A, int ldA,
                               const bf16_t* __restrict__ Bt, int K,
                               float* __restrict__ C, int ldC, int NstoreMax,
                               const float* __restrict__ dinv,
                               float* __restrict__ agg, int totalWaves) {
  int wave = (blockIdx.x * blockDim.x + threadIdx.x) >> 5;
  if (wave >= totalWaves) return;                 // wave-uniform: EXEC stays all-1s
  int lane  = threadIdx.x & 31;
  int mtile = wave;
  int hi16  = (lane >> 4) & 1;
  int lan15 = lane & 15;
  int m     = mtile * 16 + lan15;
  int kbA   = hi16 ? 8 : 0;                       // A: K-halves split by lane[4]
  int kbB   = hi16 ? 8 : 0;                       // B: K-halves split by lane[4]

  const bf16_t* arow = A + (size_t)m * ldA;

  v8f acc[NT] = {};
  for (int k0 = 0; k0 < K; k0 += 32) {
    // A fragment: elems 0..7 = K kbA..kbA+7 ; elems 8..15 = K kbA+16..kbA+23
    v8bf a0 = *(const v8bf*)(arow + k0 + kbA);
    v8bf a1 = *(const v8bf*)(arow + k0 + kbA + 16);
    v16bf av = __builtin_shufflevector(a0, a1, 0,1,2,3,4,5,6,7,8,9,10,11,12,13,14,15);
#pragma unroll
    for (int t = 0; t < NT; ++t) {
      const bf16_t* brow = Bt + (size_t)(t * 16 + lan15) * K;
      v8bf b0 = *(const v8bf*)(brow + k0 + kbB);
      v8bf b1 = *(const v8bf*)(brow + k0 + kbB + 16);
      v16bf bv = __builtin_shufflevector(b0, b1, 0,1,2,3,4,5,6,7,8,9,10,11,12,13,14,15);
      acc[t] = __builtin_amdgcn_wmma_f32_16x16x32_bf16(false, av, false, bv,
                                                       (short)0, acc[t], false, false);
    }
  }

  // D layout: VGPR r holds M = r (+8 for lanes 16..31), N = lane&15 (+16*t)
  int rbase = mtile * 16 + hi16 * 8;
#pragma unroll
  for (int t = 0; t < NT; ++t) {
    int n = t * 16 + lan15;
    if (n < NstoreMax) {
#pragma unroll
      for (int r = 0; r < 8; ++r)
        C[(size_t)(rbase + r) * ldC + n] = acc[t][r];
    }
  }
  if (agg) {                                      // ldC == HH == 64 in this path
#pragma unroll
    for (int r = 0; r < 8; ++r) {
      float d  = dinv[rbase + r];
      float d2 = d * d;
#pragma unroll
      for (int t = 0; t < NT; ++t)
        agg[(size_t)(rbase + r) * HH + t * 16 + lan15] = acc[t][r] * d2;
    }
  }
}

// ----------------------------- aggregation ---------------------------------

// One thread per (edge, feature-pair): coalesced float2 gathers from h[row],
// two native f32 atomic adds into agg[col]. Wave-uniform edge index.
__global__ void agg_edges(const float* __restrict__ h, const int* __restrict__ row,
                          const int* __restrict__ col, const float* __restrict__ norm,
                          float* __restrict__ agg) {
  int idx = blockIdx.x * blockDim.x + threadIdx.x;
  if (idx >= EE * (HH / 2)) return;               // 25.6M threads
  int e  = idx >> 5;
  int f2 = (idx & 31) << 1;
  float2 hv = *(const float2*)(h + (size_t)row[e] * HH + f2);
  float  nr = norm[e];
  float* dst = agg + (size_t)col[e] * HH + f2;
  gatomic_add_f32(dst,     hv.x * nr);
  gatomic_add_f32(dst + 1, hv.y * nr);
}

// bias + ReLU, write bf16 into JK-concat block l (also next layer's input)
__global__ void bias_relu_store(const float* __restrict__ agg, const float* __restrict__ bias,
                                bf16_t* __restrict__ concat, int l) {
  int idx = blockIdx.x * blockDim.x + threadIdx.x;
  if (idx >= NN * HH) return;
  int i = idx >> 6;
  int f = idx & 63;
  float v = agg[idx] + bias[f];
  v = fmaxf(v, 0.0f);
  concat[(size_t)i * JK + l * HH + f] = (bf16_t)v;
}

// ----------------------------- log-softmax ---------------------------------

__global__ void logsoftmax_k(const float* __restrict__ z, const float* __restrict__ bout,
                             float* __restrict__ out) {
  int i = blockIdx.x * blockDim.x + threadIdx.x;
  if (i >= NN) return;
  float v[CC];
  float m = -INFINITY;
#pragma unroll
  for (int j = 0; j < CC; ++j) {
    v[j] = z[(size_t)i * CC + j] + bout[j];
    m = fmaxf(m, v[j]);
  }
  float s = 0.0f;
#pragma unroll
  for (int j = 0; j < CC; ++j) s += __expf(v[j] - m);
  float ls = m + __logf(s);
#pragma unroll
  for (int j = 0; j < CC; ++j) out[(size_t)i * CC + j] = v[j] - ls;
}

// ----------------------------- host launcher -------------------------------

extern "C" void kernel_launch(void* const* d_in, const int* in_sizes, int n_in,
                              void* d_out, int out_size, void* d_ws, size_t ws_size,
                              hipStream_t stream) {
  const float* x    = (const float*)d_in[0];
  const int*   ei   = (const int*)  d_in[1];   // [2, E]
  const float* ew   = (const float*)d_in[2];
  const float* W0   = (const float*)d_in[3];   // [F, H]
  const float* b0   = (const float*)d_in[4];   // [H]
  const float* Wh   = (const float*)d_in[5];   // [L-1, H, H]
  const float* bh   = (const float*)d_in[6];   // [L-1, H]
  const float* Wout = (const float*)d_in[7];   // [JK, C]
  const float* bout = (const float*)d_in[8];   // [C]
  float* out = (float*)d_out;

  const int* row = ei;
  const int* col = ei + EE;

  // ---- workspace carve-out (256B aligned) ----
  char* ws = (char*)d_ws;
  size_t off = 0;
  auto take = [&](size_t bytes) -> char* {
    char* p = ws + off;
    off = (off + bytes + 255) & ~(size_t)255;
    return p;
  };
  float*  dinv   = (float*) take((size_t)NN * 4);          // deg -> dinv in place
  float*  norm   = (float*) take((size_t)EE * 4);
  bf16_t* xb     = (bf16_t*)take((size_t)NN * FF * 2);
  bf16_t* concat = (bf16_t*)take((size_t)NN * JK * 2);     // bf16 JK buffer
  float*  h      = (float*) take((size_t)NN * HH * 4);     // GEMM out (pre-agg)
  float*  agg    = (float*) take((size_t)NN * HH * 4);
  float*  z      = (float*) take((size_t)NN * CC * 4);
  bf16_t* w0t    = (bf16_t*)take((size_t)HH * FF * 2);     // [64 x 128]
  bf16_t* wht    = (bf16_t*)take((size_t)(LL - 1) * HH * HH * 2);
  bf16_t* woutt  = (bf16_t*)take((size_t)CP * JK * 2);     // [48 x 384], padded
  (void)ws_size; (void)in_sizes; (void)n_in; (void)out_size;

  const int T = 256;

  // ---- one-time prep: conversions + symmetric norm ----
  fill_zero_f32<<<(NN + T - 1) / T, T, 0, stream>>>(dinv, NN);
  cvt_f32_bf16 <<<(NN * FF + T - 1) / T, T, 0, stream>>>(x, xb, NN * FF);
  cvt_wT_bf16  <<<(FF * HH + T - 1) / T, T, 0, stream>>>(W0, w0t, FF, HH, HH);
  for (int i = 0; i < LL - 1; ++i)
    cvt_wT_bf16<<<(HH * HH + T - 1) / T, T, 0, stream>>>(Wh + (size_t)i * HH * HH,
                                                         wht + (size_t)i * HH * HH,
                                                         HH, HH, HH);
  cvt_wT_bf16<<<(JK * CP + T - 1) / T, T, 0, stream>>>(Wout, woutt, JK, CC, CP);

  deg_accum    <<<(EE + T - 1) / T, T, 0, stream>>>(col, ew, dinv);
  dinv_finalize<<<(NN + T - 1) / T, T, 0, stream>>>(dinv);
  norm_compute <<<(EE + T - 1) / T, T, 0, stream>>>(row, col, ew, dinv, norm);

  // ---- GCN layers ----
  const int Mtiles      = NN / 16;                       // 3125 waves per GEMM
  const int gemmBlocks  = (Mtiles + 3) / 4;              // 128 threads = 4 waves
  const int aggN        = (NN * HH + T - 1) / T;
  const int edgeBlocks  = (EE * (HH / 2) + T - 1) / T;   // 100000 blocks

  for (int l = 0; l < LL; ++l) {
    const bf16_t* A   = (l == 0) ? xb : (concat + (size_t)(l - 1) * HH);
    const int     ldA = (l == 0) ? FF : JK;
    const int     K   = (l == 0) ? FF : HH;
    const bf16_t* Bt  = (l == 0) ? w0t : (wht + (size_t)(l - 1) * HH * HH);
    const float*  bias= (l == 0) ? b0  : (bh + (size_t)(l - 1) * HH);

    gemm_bf16_wmma<4><<<gemmBlocks, 128, 0, stream>>>(A, ldA, Bt, K, h, HH, HH,
                                                      dinv, agg, Mtiles);
    agg_edges<<<edgeBlocks, T, 0, stream>>>(h, row, col, norm, agg);
    bias_relu_store<<<aggN, T, 0, stream>>>(agg, bias, concat, l);
  }

  // ---- JK output GEMM: [N,384] x [384,40] (N padded to 48, masked store) ----
  gemm_bf16_wmma<3><<<gemmBlocks, 128, 0, stream>>>(concat, JK, woutt, JK, z, CC, CC,
                                                    nullptr, nullptr, Mtiles);

  // ---- log-softmax ----
  logsoftmax_k<<<(NN + T - 1) / T, T, 0, stream>>>(z, bout, out);
}
